// ImageCaptioningModel_6768868458845
// MI455X (gfx1250) — matrix operand order repaired and checked by reading, not compiled, not run on hardware
//
#include <hip/hip_runtime.h>
#include <hip/hip_bf16.h>

// ---------------- model dimensions ----------------
#define BZ 16
#define TT 64
#define VV 32000
#define CC 768
#define NHH 12
#define HSS 64
#define LL 4
#define NP 196          // num patches
#define PDD 768         // patch dim = 3*16*16
#define FF 3072         // 4*C

// ---------------- CDNA5 async-copy availability ----------------
#define AS1 __attribute__((address_space(1)))
#define AS3 __attribute__((address_space(3)))
#if defined(__has_builtin)
# if __has_builtin(__builtin_amdgcn_global_load_async_to_lds_b128)
#  define HAVE_ASYNC_LDS 1
# endif
#endif
#ifndef HAVE_ASYNC_LDS
# define HAVE_ASYNC_LDS 0
#endif

__device__ __forceinline__ void wait_async0() {
#if HAVE_ASYNC_LDS
# if __has_builtin(__builtin_amdgcn_s_wait_asynccnt)
    __builtin_amdgcn_s_wait_asynccnt(0);
# else
    asm volatile("s_wait_asynccnt 0x0" ::: "memory");
# endif
#endif
}

#if HAVE_ASYNC_LDS
typedef __attribute__((__vector_size__(16))) int v4i_t;
// 16B async copy HBM -> LDS (no VGPR round trip), tracked by ASYNCcnt
__device__ __forceinline__ void async_copy16(const unsigned short* gsrc, unsigned short* ldst) {
    v4i_t* g = reinterpret_cast<v4i_t*>(const_cast<unsigned short*>(gsrc));
    v4i_t* l = reinterpret_cast<v4i_t*>(ldst);
    __builtin_amdgcn_global_load_async_to_lds_b128((AS1 v4i_t*)g, (AS3 v4i_t*)l, 0, 0);
}
#endif

// ---------------- WMMA types ----------------
typedef __attribute__((ext_vector_type(16))) __bf16 v16bf;
typedef __attribute__((ext_vector_type(8)))  float  v8f;

union Frag16 { v16bf v; uint4 q[2]; };

__device__ __forceinline__ unsigned short f32_to_bf16u(float f) {
    unsigned int u = __float_as_uint(f);
    u += 0x7fffu + ((u >> 16) & 1u);     // RNE
    return (unsigned short)(u >> 16);
}
__device__ __forceinline__ unsigned int pack_bf16x2(float a, float b) {
    unsigned int ua = __float_as_uint(a);
    unsigned int ub = __float_as_uint(b);
    ua += 0x7fffu + ((ua >> 16) & 1u);
    ub += 0x7fffu + ((ub >> 16) & 1u);
    return (ua >> 16) | (ub & 0xffff0000u);
}

// ---------------- generic batched GEMM (bf16 in, f32 acc) ----------------
// For z = blockIdx.z: zo = z / zdiv, zi = z % zdiv.
// A addr = A + zo*aso + zi*asi + m*asr + k            (A k-stride fixed to 1)
// B addr = B + zo*bso + zi*bsi + k*bsr + n*bsc
// D addr = D + zo*dso + zi*dsi + m*dsr + n            (resid uses D addressing, f32 only)
struct GP {
    const unsigned short* A; const unsigned short* B;
    const float* bias; const float* resid;
    void* D;
    int M, N, K, zdiv;
    long long aso, asi; int asr;
    long long bso, bsi; int bsr, bsc;
    long long dso, dsi; int dsr;
    float alpha;
    int act;     // 0 = none, 1 = exact GELU
    int outBf;   // 0 = f32 D, 1 = bf16 D
};

// 256 threads = 8 wave32s. Block tile 64(M) x 64(N), K-step 32, LDS double-buffered.
// Wave w owns two 16x16 D tiles: (tm, tn) and (tm+32, tn), tm=(w>>2)*16, tn=(w&3)*16.
// KFAST: K % 32 == 0 (vector/async staging). BPATH: 1 = B k-contiguous (bsr==1),
// 2 = B n-contiguous (bsc==1, N%64==0), 0 = generic strided.
template<bool KFAST, int BPATH>
__global__ __launch_bounds__(256) void wmma_gemm_kernel(GP p) {
    // rows are 40 ushorts = 80B (multiple of 16B) so b128 LDS ops stay aligned
    __shared__ __align__(16) unsigned short As[2][64][40];   // [buf][m][k]
    __shared__ __align__(16) unsigned short Bs[2][64][40];   // [buf][n][k] (B transposed)

    const int tid  = threadIdx.x;
    const int wave = tid >> 5;
    const int lane = tid & 31;
    const int lm   = lane & 15;
    const int hi   = lane >> 4;

    const int m0 = blockIdx.x * 64;
    const int n0 = blockIdx.y * 64;
    const int z  = blockIdx.z;
    const int zo = z / p.zdiv;
    const int zi = z % p.zdiv;

    const unsigned short* Ab = p.A + zo * p.aso + zi * p.asi;
    const unsigned short* Bb = p.B + zo * p.bso + zi * p.bsi;

    const int tm = (wave >> 2) * 16;
    const int tn = (wave & 3) * 16;

    v8f acc0 = {0.f, 0.f, 0.f, 0.f, 0.f, 0.f, 0.f, 0.f};
    v8f acc1 = {0.f, 0.f, 0.f, 0.f, 0.f, 0.f, 0.f, 0.f};

    // staging assignments
    const int arow = tid >> 2;            // 0..63  (A row)
    const int akb  = (tid & 3) * 8;       // 0,8,16,24
    const int bn   = tid & 63;            // 0..63  (B col, k-strided paths)
    const int bkb  = (tid >> 6) * 8;      // 0,8,16,24
    const int bk2  = tid >> 3;            // 0..31  (B row, n-contiguous path)
    const int bnb2 = (tid & 7) * 8;       // 0..56

    const bool aInterior = (m0 + 64 <= p.M);
    const bool bInterior = (n0 + 64 <= p.N);

    auto stageA = [&](int buf, int kk) {
        if constexpr (KFAST) {
            const int gm = m0 + arow;
#if HAVE_ASYNC_LDS
            if (aInterior) {   // pure bf16 copy: HBM -> LDS, no VGPR round-trip
                async_copy16(Ab + (long long)gm * p.asr + kk + akb, &As[buf][arow][akb]);
                return;
            }
#endif
            const int cgm = min(gm, p.M - 1);
            const unsigned short* ap = Ab + (long long)cgm * p.asr + kk + akb;
            uint4 q = *reinterpret_cast<const uint4*>(ap);
            if (gm >= p.M) { q.x = 0u; q.y = 0u; q.z = 0u; q.w = 0u; }
            *reinterpret_cast<uint4*>(&As[buf][arow][akb]) = q;
        } else {
            const int gm  = m0 + arow;
            const int cgm = min(gm, p.M - 1);
            const unsigned short* ap = Ab + (long long)cgm * p.asr;
            unsigned int pk[4];
            #pragma unroll
            for (int j = 0; j < 4; ++j) {
                const int gk0 = kk + akb + 2 * j, gk1 = gk0 + 1;
                unsigned int u0 = ap[min(gk0, p.K - 1)];
                unsigned int u1 = ap[min(gk1, p.K - 1)];
                u0 = (gm < p.M && gk0 < p.K) ? u0 : 0u;
                u1 = (gm < p.M && gk1 < p.K) ? u1 : 0u;
                pk[j] = u0 | (u1 << 16);
            }
            uint4 q; q.x = pk[0]; q.y = pk[1]; q.z = pk[2]; q.w = pk[3];
            *reinterpret_cast<uint4*>(&As[buf][arow][akb]) = q;
        }
    };

    auto stageB = [&](int buf, int kk) {
        if constexpr (BPATH == 1) {          // k-contiguous (q @ k^T): pure copy
            const int gn = n0 + bn;
#if HAVE_ASYNC_LDS
            if (bInterior) {
                async_copy16(Bb + (long long)gn * p.bsc + kk + bkb, &Bs[buf][bn][bkb]);
                return;
            }
#endif
            const int cgn = min(gn, p.N - 1);
            const unsigned short* bp = Bb + (long long)cgn * p.bsc + kk + bkb;
            uint4 q = *reinterpret_cast<const uint4*>(bp);
            if (gn >= p.N) { q.x = 0u; q.y = 0u; q.z = 0u; q.w = 0u; }
            *reinterpret_cast<uint4*>(&Bs[buf][bn][bkb]) = q;
        } else if constexpr (BPATH == 2) {   // n-contiguous weights: transpose scatter
            const int gk = kk + bk2;
            const unsigned short* bp = Bb + (long long)gk * p.bsr + n0 + bnb2;
            uint4 q = *reinterpret_cast<const uint4*>(bp);
            const unsigned short* e = reinterpret_cast<const unsigned short*>(&q);
            #pragma unroll
            for (int j = 0; j < 8; ++j) Bs[buf][bnb2 + j][bk2] = e[j];
        } else {                              // generic strided, clamped
            const int gn  = n0 + bn;
            const int cgn = min(gn, p.N - 1);
            const unsigned short* bp = Bb + (long long)cgn * p.bsc;
            unsigned int pk[4];
            #pragma unroll
            for (int j = 0; j < 4; ++j) {
                const int gk0 = kk + bkb + 2 * j, gk1 = gk0 + 1;
                unsigned int u0 = bp[(long long)min(gk0, p.K - 1) * p.bsr];
                unsigned int u1 = bp[(long long)min(gk1, p.K - 1) * p.bsr];
                u0 = (gn < p.N && gk0 < p.K) ? u0 : 0u;
                u1 = (gn < p.N && gk1 < p.K) ? u1 : 0u;
                pk[j] = u0 | (u1 << 16);
            }
            uint4 q; q.x = pk[0]; q.y = pk[1]; q.z = pk[2]; q.w = pk[3];
            *reinterpret_cast<uint4*>(&Bs[buf][bn][bkb]) = q;
        }
    };

    stageA(0, 0);
    stageB(0, 0);
    int buf = 0;
    for (int kk = 0; kk < p.K; kk += 32) {
        wait_async0();          // drain this wave's async LDS writes before publishing
        __syncthreads();
        if (kk + 32 < p.K) {
            stageA(buf ^ 1, kk + 32);
            stageB(buf ^ 1, kk + 32);
            if (KFAST && kk + 64 < p.K) {   // prefetch the tile after next
                __builtin_prefetch(Ab + (long long)min(m0 + arow, p.M - 1) * p.asr + kk + 64 + akb, 0, 3);
                __builtin_prefetch(Bb + (long long)(kk + 64) * p.bsr
                                      + (long long)min(n0 + bn, p.N - 1) * p.bsc, 0, 3);
            }
        }
        // ---- fragments per ISA wave32 16-bit WMMA layouts ----
        Frag16 fa0, fa1, fb;
        fa0.q[0] = *reinterpret_cast<const uint4*>(&As[buf][tm + lm][8 * hi]);
        fa0.q[1] = *reinterpret_cast<const uint4*>(&As[buf][tm + lm][16 + 8 * hi]);
        fa1.q[0] = *reinterpret_cast<const uint4*>(&As[buf][tm + 32 + lm][8 * hi]);
        fa1.q[1] = *reinterpret_cast<const uint4*>(&As[buf][tm + 32 + lm][16 + 8 * hi]);
        fb.q[0]  = *reinterpret_cast<const uint4*>(&Bs[buf][tn + lm][8 * hi]);
        fb.q[1]  = *reinterpret_cast<const uint4*>(&Bs[buf][tn + lm][16 + 8 * hi]);

        acc0 = __builtin_amdgcn_wmma_f32_16x16x32_bf16(
            false, fa0.v, false, fb.v, (short)0, acc0, false, false);
        acc1 = __builtin_amdgcn_wmma_f32_16x16x32_bf16(
            false, fa1.v, false, fb.v, (short)0, acc1, false, false);
        buf ^= 1;
    }

    // ---- epilogue ----
    const int gn = n0 + tn + lm;
    if (gn < p.N) {
        const float bv = p.bias ? p.bias[gn] : 0.f;
        if (p.outBf) {
            unsigned short* Db = (unsigned short*)p.D + zo * p.dso + zi * p.dsi;
            #pragma unroll
            for (int r = 0; r < 8; ++r) {
                const int gm = m0 + tm + r + 8 * hi;
                if (gm < p.M) {
                    float v = acc0[r] * p.alpha + bv;
                    if (p.act == 1) v = 0.5f * v * (1.f + erff(v * 0.70710678118654752f));
                    Db[(long long)gm * p.dsr + gn] = f32_to_bf16u(v);
                }
            }
            #pragma unroll
            for (int r = 0; r < 8; ++r) {
                const int gm = m0 + 32 + tm + r + 8 * hi;
                if (gm < p.M) {
                    float v = acc1[r] * p.alpha + bv;
                    if (p.act == 1) v = 0.5f * v * (1.f + erff(v * 0.70710678118654752f));
                    Db[(long long)gm * p.dsr + gn] = f32_to_bf16u(v);
                }
            }
        } else {
            float* Db = (float*)p.D + zo * p.dso + zi * p.dsi;
            const float* Rb = p.resid ? (p.resid + zo * p.dso + zi * p.dsi) : nullptr;
            #pragma unroll
            for (int r = 0; r < 8; ++r) {
                const int gm = m0 + tm + r + 8 * hi;
                if (gm < p.M) {
                    float v = acc0[r] * p.alpha + bv;
                    if (p.act == 1) v = 0.5f * v * (1.f + erff(v * 0.70710678118654752f));
                    const long long off = (long long)gm * p.dsr + gn;
                    if (Rb) v += Rb[off];
                    Db[off] = v;
                }
            }
            #pragma unroll
            for (int r = 0; r < 8; ++r) {
                const int gm = m0 + 32 + tm + r + 8 * hi;
                if (gm < p.M) {
                    float v = acc1[r] * p.alpha + bv;
                    if (p.act == 1) v = 0.5f * v * (1.f + erff(v * 0.70710678118654752f));
                    const long long off = (long long)gm * p.dsr + gn;
                    if (Rb) v += Rb[off];
                    Db[off] = v;
                }
            }
        }
    }
}

// ---------------- f32 -> bf16 converter (vectorized x4) ----------------
__global__ __launch_bounds__(256) void cvt_bf16_kernel(const float* __restrict__ in,
                                                       unsigned short* __restrict__ out,
                                                       int total4) {
    int i = blockIdx.x * 256 + threadIdx.x;
    if (i >= total4) return;
    float4 f = reinterpret_cast<const float4*>(in)[i];
    uint2 o;
    o.x = pack_bf16x2(f.x, f.y);
    o.y = pack_bf16x2(f.z, f.w);
    reinterpret_cast<uint2*>(out)[i] = o;
}

// ---------------- LayerNorm (one row per block, bf16 out) ----------------
__global__ __launch_bounds__(256) void ln_kernel(const float* __restrict__ x,
                                                 const float* __restrict__ w,
                                                 const float* __restrict__ b,
                                                 unsigned short* __restrict__ out, int C) {
    __shared__ float red[256];
    const int row = blockIdx.x, tid = threadIdx.x;
    const float* xr = x + (long long)row * C;
    float s = 0.f;
    for (int c = tid; c < C; c += 256) s += xr[c];
    red[tid] = s; __syncthreads();
    for (int o = 128; o > 0; o >>= 1) { if (tid < o) red[tid] += red[tid + o]; __syncthreads(); }
    const float mean = red[0] / C;
    __syncthreads();
    float s2 = 0.f;
    for (int c = tid; c < C; c += 256) { float d = xr[c] - mean; s2 += d * d; }
    red[tid] = s2; __syncthreads();
    for (int o = 128; o > 0; o >>= 1) { if (tid < o) red[tid] += red[tid + o]; __syncthreads(); }
    const float rstd = rsqrtf(red[0] / C + 1e-5f);
    unsigned short* orow = out + (long long)row * C;
    for (int c = tid; c < C; c += 256)
        orow[c] = f32_to_bf16u((xr[c] - mean) * rstd * w[c] + b[c]);
}

// ---------------- softmax: f32 scores in -> bf16 probs out; S <= 256 ----------------
__global__ __launch_bounds__(256) void softmax_kernel(const float* __restrict__ in,
                                                      unsigned short* __restrict__ out,
                                                      int T, int S, int causal) {
    __shared__ float red[256];
    const int row = blockIdx.x, tid = threadIdx.x;
    const int t = row % T;
    const int valid = causal ? (t + 1) : S;
    const float* r = in + (long long)row * S;
    float v = (tid < valid) ? r[tid] : -3.4e38f;
    red[tid] = v; __syncthreads();
    for (int o = 128; o > 0; o >>= 1) { if (tid < o) red[tid] = fmaxf(red[tid], red[tid + o]); __syncthreads(); }
    const float mx = red[0];
    __syncthreads();
    const float e = (tid < valid) ? __expf(v - mx) : 0.f;
    red[tid] = e; __syncthreads();
    for (int o = 128; o > 0; o >>= 1) { if (tid < o) red[tid] += red[tid + o]; __syncthreads(); }
    const float inv = 1.f / red[0];
    if (tid < S) out[(long long)row * S + tid] = f32_to_bf16u(e * inv);
}

// ---------------- patchify: img[B,3,224,224] -> xp[B,196,768] (bf16) ----------------
__global__ __launch_bounds__(256) void patchify_kernel(const float* __restrict__ img,
                                                       unsigned short* __restrict__ xp, int total) {
    int i = blockIdx.x * 256 + threadIdx.x;
    if (i >= total) return;
    const int d = i % PDD;
    const int bp = i / PDD;
    const int p = bp % NP;
    const int b = bp / NP;
    const int c = d / 256;
    const int rr = d % 256;
    const int ii = rr / 16, jj = rr % 16;
    const int py = p / 14, px = p % 14;
    const int h = py * 16 + ii, w = px * 16 + jj;
    xp[i] = f32_to_bf16u(img[(((long long)b * 3 + c) * 224 + h) * 224 + w]);
}

__global__ __launch_bounds__(256) void addpos_kernel(float* x, const float* __restrict__ pos, int total) {
    int i = blockIdx.x * 256 + threadIdx.x;
    if (i >= total) return;
    const int c = i % CC;
    const int m = i / CC;
    const int p = m % NP;
    x[i] += pos[p * CC + c];
}

__global__ __launch_bounds__(256) void embed_kernel(const int* __restrict__ tok,
                                                    const float* __restrict__ te,
                                                    const float* __restrict__ pe,
                                                    float* __restrict__ y, int total) {
    int i = blockIdx.x * 256 + threadIdx.x;
    if (i >= total) return;
    const int c = i % CC;
    const int bt = i / CC;
    const int t = bt % TT;
    y[i] = te[(long long)tok[bt] * CC + c] + pe[t * CC + c];
}

// ---------------- host-side helpers ----------------
static inline void launch_gemm(hipStream_t st, const unsigned short* A, const unsigned short* B,
                               const float* bias, const float* resid, void* D,
                               int M, int N, int K, int Z, int zdiv,
                               long long aso, long long asi, int asr,
                               long long bso, long long bsi, int bsr, int bsc,
                               long long dso, long long dsi, int dsr,
                               float alpha, int act, int outBf) {
    GP p;
    p.A = A; p.B = B; p.bias = bias; p.resid = resid; p.D = D;
    p.M = M; p.N = N; p.K = K; p.zdiv = zdiv;
    p.aso = aso; p.asi = asi; p.asr = asr;
    p.bso = bso; p.bsi = bsi; p.bsr = bsr; p.bsc = bsc;
    p.dso = dso; p.dsi = dsi; p.dsr = dsr;
    p.alpha = alpha; p.act = act; p.outBf = outBf;
    dim3 grid((M + 63) / 64, (N + 63) / 64, Z);
    const dim3 blk(256);
    const bool kfast = (K & 31) == 0;
    if (kfast && bsr == 1)
        wmma_gemm_kernel<true, 1><<<grid, blk, 0, st>>>(p);
    else if (kfast && bsc == 1 && (N & 63) == 0)
        wmma_gemm_kernel<true, 2><<<grid, blk, 0, st>>>(p);
    else if (kfast)
        wmma_gemm_kernel<true, 0><<<grid, blk, 0, st>>>(p);
    else
        wmma_gemm_kernel<false, 0><<<grid, blk, 0, st>>>(p);
}

static inline void cvt_bf16(hipStream_t st, const float* in, unsigned short* out, long long total) {
    const int t4 = (int)(total >> 2);
    cvt_bf16_kernel<<<(t4 + 255) / 256, 256, 0, st>>>(in, out, t4);
}

extern "C" void kernel_launch(void* const* d_in, const int* in_sizes, int n_in,
                              void* d_out, int out_size, void* d_ws, size_t ws_size,
                              hipStream_t stream) {
    (void)in_sizes; (void)n_in; (void)out_size; (void)ws_size;

    const float* img      = (const float*)d_in[0];
    const int*   tok      = (const int*)  d_in[1];
    const float* patch_w  = (const float*)d_in[2];
    const float* patch_b  = (const float*)d_in[3];
    const float* patch_pos= (const float*)d_in[4];
    const float* tok_emb  = (const float*)d_in[5];
    const float* pos_emb  = (const float*)d_in[6];
    const float* e_ln1w = (const float*)d_in[7],  *e_ln1b = (const float*)d_in[8];
    const float* e_Wq   = (const float*)d_in[9],  *e_Wk   = (const float*)d_in[10];
    const float* e_Wv   = (const float*)d_in[11], *e_Wo   = (const float*)d_in[12];
    const float* e_bo   = (const float*)d_in[13];
    const float* e_ln2w = (const float*)d_in[14], *e_ln2b = (const float*)d_in[15];
    const float* e_f1w  = (const float*)d_in[16], *e_f1b  = (const float*)d_in[17];
    const float* e_f2w  = (const float*)d_in[18], *e_f2b  = (const float*)d_in[19];
    const float* d_ln1w = (const float*)d_in[20], *d_ln1b = (const float*)d_in[21];
    const float* d_sWq  = (const float*)d_in[22], *d_sWk  = (const float*)d_in[23];
    const float* d_sWv  = (const float*)d_in[24], *d_sWo  = (const float*)d_in[25];
    const float* d_sbo  = (const float*)d_in[26];
    const float* d_ln2w = (const float*)d_in[27], *d_ln2b = (const float*)d_in[28];
    const float* d_cWq  = (const float*)d_in[29], *d_cWk  = (const float*)d_in[30];
    const float* d_cWv  = (const float*)d_in[31], *d_cWo  = (const float*)d_in[32];
    const float* d_cbo  = (const float*)d_in[33];
    const float* d_ln3w = (const float*)d_in[34], *d_ln3b = (const float*)d_in[35];
    const float* d_f1w  = (const float*)d_in[36], *d_f1b  = (const float*)d_in[37];
    const float* d_f2w  = (const float*)d_in[38], *d_f2b  = (const float*)d_in[39];
    const float* lnf_w  = (const float*)d_in[40], *lnf_b  = (const float*)d_in[41];
    const float* lm_w   = (const float*)d_in[42], *lm_b   = (const float*)d_in[43];
    float* logits = (float*)d_out;

    // ---- workspace layout ----
    const long long ME = (long long)BZ * NP;   // 3136 encoder rows
    const long long MD = (long long)BZ * TT;   // 1024 decoder rows
    const long long XE = ME * CC;              // 2408448
    const long long YE = MD * CC;              // 786432
    const long long SE = (long long)BZ * NHH * NP * NP;   // 7375872 (max scores)
    const long long HE = ME * FF;              // 9633792

    float* ws = (float*)d_ws;
    float* X  = ws;                 // enc residual stream / enc_out (f32)
    float* Y  = X + XE;             // dec residual stream (f32)
    float* Sc = Y + YE;             // raw attention scores (f32)
    unsigned short* us  = (unsigned short*)(Sc + SE);
    unsigned short* LNb = us;             // layernorm output (bf16)
    unsigned short* Qb  = LNb + XE;
    unsigned short* Kbb = Qb  + XE;
    unsigned short* Vbb = Kbb + XE;
    unsigned short* Ob  = Vbb + XE;       // attention out (head-concat, bf16)
    unsigned short* Hb  = Ob  + XE;       // mlp hidden (bf16)
    unsigned short* XPb = Hb  + HE;       // patchified image (bf16)
    unsigned short* Xbf = XPb + XE;       // bf16 copy of enc_out
    unsigned short* ScB = Xbf + XE;       // softmax probs (bf16)
    unsigned short* WBu = ScB + SE;       // reusable weight scratch (bf16, max = lm_w)

    // strides
    const long long WQZ = (long long)CC * HSS;        // 49152: per-head weight block
    const long long WL  = (long long)NHH * CC * HSS;  // 589824: per-layer attn weights
    const long long FCL = (long long)CC * FF;         // 2359296: per-layer fc weights
    const long long ETC = (long long)NP * CC;         // 150528: enc per-batch act stride
    const long long DTC = (long long)TT * CC;         // 49152:  dec per-batch act stride

    // ================= encoder =================
    {
        const int tot = (int)(ME * PDD);
        patchify_kernel<<<(tot + 255) / 256, 256, 0, stream>>>(img, XPb, tot);
    }
    cvt_bf16(stream, patch_w, WBu, (long long)PDD * CC);
    launch_gemm(stream, XPb, WBu, patch_b, nullptr, X,
                (int)ME, CC, PDD, 1, 1,
                0, 0, PDD, 0, 0, CC, 1, 0, 0, CC, 1.f, 0, 0);
    {
        const int tot = (int)XE;
        addpos_kernel<<<(tot + 255) / 256, 256, 0, stream>>>(X, patch_pos, tot);
    }

    for (int l = 0; l < LL; ++l) {
        ln_kernel<<<(int)ME, 256, 0, stream>>>(X, e_ln1w + l * CC, e_ln1b + l * CC, LNb, CC);
        // per-head Q/K/V projections: z = head, bf16 out
        cvt_bf16(stream, e_Wq + l * WL, WBu, WL);
        launch_gemm(stream, LNb, WBu, nullptr, nullptr, Qb,
                    (int)ME, HSS, CC, NHH, 1,
                    0, 0, CC, WQZ, 0, HSS, 1, HSS, 0, CC, 1.f, 0, 1);
        cvt_bf16(stream, e_Wk + l * WL, WBu, WL);
        launch_gemm(stream, LNb, WBu, nullptr, nullptr, Kbb,
                    (int)ME, HSS, CC, NHH, 1,
                    0, 0, CC, WQZ, 0, HSS, 1, HSS, 0, CC, 1.f, 0, 1);
        cvt_bf16(stream, e_Wv + l * WL, WBu, WL);
        launch_gemm(stream, LNb, WBu, nullptr, nullptr, Vbb,
                    (int)ME, HSS, CC, NHH, 1,
                    0, 0, CC, WQZ, 0, HSS, 1, HSS, 0, CC, 1.f, 0, 1);
        // scores = q @ k^T * 1/8 ; z = b*NH + h (f32 out)
        launch_gemm(stream, Qb, Kbb, nullptr, nullptr, Sc,
                    NP, NP, HSS, BZ * NHH, NHH,
                    ETC, HSS, CC,
                    ETC, HSS, 1, CC,
                    (long long)NHH * NP * NP, (long long)NP * NP, NP,
                    0.125f, 0, 0);
        softmax_kernel<<<BZ * NHH * NP, 256, 0, stream>>>(Sc, ScB, NP, NP, 0);
        // o = a @ v (bf16 out)
        launch_gemm(stream, ScB, Vbb, nullptr, nullptr, Ob,
                    NP, HSS, NP, BZ * NHH, NHH,
                    (long long)NHH * NP * NP, (long long)NP * NP, NP,
                    ETC, HSS, CC, 1,
                    ETC, HSS, CC, 1.f, 0, 1);
        // projection + bias + residual (f32, in place on X)
        cvt_bf16(stream, e_Wo + l * WL, WBu, WL);
        launch_gemm(stream, Ob, WBu, e_bo + l * CC, X, X,
                    (int)ME, CC, CC, 1, 1,
                    0, 0, CC, 0, 0, CC, 1, 0, 0, CC, 1.f, 0, 0);
        // MLP
        ln_kernel<<<(int)ME, 256, 0, stream>>>(X, e_ln2w + l * CC, e_ln2b + l * CC, LNb, CC);
        cvt_bf16(stream, e_f1w + l * FCL, WBu, FCL);
        launch_gemm(stream, LNb, WBu, e_f1b + l * FF, nullptr, Hb,
                    (int)ME, FF, CC, 1, 1,
                    0, 0, CC, 0, 0, FF, 1, 0, 0, FF, 1.f, 1, 1);
        cvt_bf16(stream, e_f2w + l * FCL, WBu, FCL);
        launch_gemm(stream, Hb, WBu, e_f2b + l * CC, X, X,
                    (int)ME, CC, FF, 1, 1,
                    0, 0, FF, 0, 0, CC, 1, 0, 0, CC, 1.f, 0, 0);
    }
    // bf16 copy of enc_out for cross-attention K/V projections
    cvt_bf16(stream, X, Xbf, XE);

    // ================= decoder =================
    {
        const int tot = (int)YE;
        embed_kernel<<<(tot + 255) / 256, 256, 0, stream>>>(tok, tok_emb, pos_emb, Y, tot);
    }

    for (int l = 0; l < LL; ++l) {
        // ---- causal self-attention ----
        ln_kernel<<<(int)MD, 256, 0, stream>>>(Y, d_ln1w + l * CC, d_ln1b + l * CC, LNb, CC);
        cvt_bf16(stream, d_sWq + l * WL, WBu, WL);
        launch_gemm(stream, LNb, WBu, nullptr, nullptr, Qb,
                    (int)MD, HSS, CC, NHH, 1, 0, 0, CC, WQZ, 0, HSS, 1, HSS, 0, CC, 1.f, 0, 1);
        cvt_bf16(stream, d_sWk + l * WL, WBu, WL);
        launch_gemm(stream, LNb, WBu, nullptr, nullptr, Kbb,
                    (int)MD, HSS, CC, NHH, 1, 0, 0, CC, WQZ, 0, HSS, 1, HSS, 0, CC, 1.f, 0, 1);
        cvt_bf16(stream, d_sWv + l * WL, WBu, WL);
        launch_gemm(stream, LNb, WBu, nullptr, nullptr, Vbb,
                    (int)MD, HSS, CC, NHH, 1, 0, 0, CC, WQZ, 0, HSS, 1, HSS, 0, CC, 1.f, 0, 1);
        launch_gemm(stream, Qb, Kbb, nullptr, nullptr, Sc,
                    TT, TT, HSS, BZ * NHH, NHH,
                    DTC, HSS, CC,
                    DTC, HSS, 1, CC,
                    (long long)NHH * TT * TT, (long long)TT * TT, TT,
                    0.125f, 0, 0);
        softmax_kernel<<<BZ * NHH * TT, 256, 0, stream>>>(Sc, ScB, TT, TT, 1);
        launch_gemm(stream, ScB, Vbb, nullptr, nullptr, Ob,
                    TT, HSS, TT, BZ * NHH, NHH,
                    (long long)NHH * TT * TT, (long long)TT * TT, TT,
                    DTC, HSS, CC, 1,
                    DTC, HSS, CC, 1.f, 0, 1);
        cvt_bf16(stream, d_sWo + l * WL, WBu, WL);
        launch_gemm(stream, Ob, WBu, d_sbo + l * CC, Y, Y,
                    (int)MD, CC, CC, 1, 1, 0, 0, CC, 0, 0, CC, 1, 0, 0, CC, 1.f, 0, 0);

        // ---- cross-attention (ctx = enc_out = Xbf, no LN on ctx) ----
        ln_kernel<<<(int)MD, 256, 0, stream>>>(Y, d_ln2w + l * CC, d_ln2b + l * CC, LNb, CC);
        cvt_bf16(stream, d_cWq + l * WL, WBu, WL);
        launch_gemm(stream, LNb, WBu, nullptr, nullptr, Qb,
                    (int)MD, HSS, CC, NHH, 1, 0, 0, CC, WQZ, 0, HSS, 1, HSS, 0, CC, 1.f, 0, 1);
        cvt_bf16(stream, d_cWk + l * WL, WBu, WL);
        launch_gemm(stream, Xbf, WBu, nullptr, nullptr, Kbb,
                    (int)ME, HSS, CC, NHH, 1, 0, 0, CC, WQZ, 0, HSS, 1, HSS, 0, CC, 1.f, 0, 1);
        cvt_bf16(stream, d_cWv + l * WL, WBu, WL);
        launch_gemm(stream, Xbf, WBu, nullptr, nullptr, Vbb,
                    (int)ME, HSS, CC, NHH, 1, 0, 0, CC, WQZ, 0, HSS, 1, HSS, 0, CC, 1.f, 0, 1);
        launch_gemm(stream, Qb, Kbb, nullptr, nullptr, Sc,
                    TT, NP, HSS, BZ * NHH, NHH,
                    DTC, HSS, CC,
                    ETC, HSS, 1, CC,
                    (long long)NHH * TT * NP, (long long)TT * NP, NP,
                    0.125f, 0, 0);
        softmax_kernel<<<BZ * NHH * TT, 256, 0, stream>>>(Sc, ScB, TT, NP, 0);
        launch_gemm(stream, ScB, Vbb, nullptr, nullptr, Ob,
                    TT, HSS, NP, BZ * NHH, NHH,
                    (long long)NHH * TT * NP, (long long)TT * NP, NP,
                    ETC, HSS, CC, 1,
                    DTC, HSS, CC, 1.f, 0, 1);
        cvt_bf16(stream, d_cWo + l * WL, WBu, WL);
        launch_gemm(stream, Ob, WBu, d_cbo + l * CC, Y, Y,
                    (int)MD, CC, CC, 1, 1, 0, 0, CC, 0, 0, CC, 1, 0, 0, CC, 1.f, 0, 0);

        // ---- MLP ----
        ln_kernel<<<(int)MD, 256, 0, stream>>>(Y, d_ln3w + l * CC, d_ln3b + l * CC, LNb, CC);
        cvt_bf16(stream, d_f1w + l * FCL, WBu, FCL);
        launch_gemm(stream, LNb, WBu, d_f1b + l * FF, nullptr, Hb,
                    (int)MD, FF, CC, 1, 1, 0, 0, CC, 0, 0, FF, 1, 0, 0, FF, 1.f, 1, 1);
        cvt_bf16(stream, d_f2w + l * FCL, WBu, FCL);
        launch_gemm(stream, Hb, WBu, d_f2b + l * CC, Y, Y,
                    (int)MD, CC, FF, 1, 1, 0, 0, FF, 0, 0, CC, 1, 0, 0, CC, 1.f, 0, 0);
    }

    // ================= head =================
    ln_kernel<<<(int)MD, 256, 0, stream>>>(Y, lnf_w, lnf_b, LNb, CC);
    cvt_bf16(stream, lm_w, WBu, (long long)CC * VV);
    launch_gemm(stream, LNb, WBu, lm_b, nullptr, logits,
                (int)MD, VV, CC, 1, 1,
                0, 0, CC, 0, 0, VV, 1, 0, 0, VV, 1.f, 0, 0);
}
